// GCFNet_26766236188937
// MI455X (gfx1250) — matrix operand order repaired and checked by reading, not compiled, not run on hardware
//
#include <hip/hip_runtime.h>
#include <hip/hip_bf16.h>

// ---------------- model dims ----------------
#define NN    4000
#define TT    60
#define FF    16
#define HH    128
#define CTX   32
#define EMB   32
#define NIND  64
#define NHEAD 4
#define HD    32
#define NT    (NN * TT)   // 240000

typedef __attribute__((ext_vector_type(16))) _Float16 v16h;
typedef __attribute__((ext_vector_type(8)))  _Float16 v8h;
typedef __attribute__((ext_vector_type(8)))  float    v8f;

__device__ __forceinline__ v8f wmma_f16(v16h a, v16h b, v8f c) {
  return __builtin_amdgcn_wmma_f32_16x16x32_f16(false, a, false, b, (short)0, c, false, false);
}

__device__ __forceinline__ float wave_red(float v) {
#pragma unroll
  for (int off = 16; off > 0; off >>= 1) v += __shfl_xor(v, off, 32);
  return v;
}

// ---------------- weight pack: f32 [K,N] row-major -> f16 [N,K] (B-operand layout) ----
__global__ void k_pack(const float* __restrict__ w, _Float16* __restrict__ out, int K, int N) {
  int idx = blockIdx.x * 256 + threadIdx.x;
  if (idx >= K * N) return;
  int n = idx / K, k = idx % K;
  out[idx] = (_Float16)w[(size_t)k * N + n];
}

// ---------------- conv1d(k=3,p=1) + GLU ----------------
__global__ void k_conv_glu(const float* __restrict__ df, const float* __restrict__ w,
                           const float* __restrict__ bias,
                           float* __restrict__ x32, _Float16* __restrict__ xh) {
  size_t idx = (size_t)blockIdx.x * 256 + threadIdx.x;   // over NT*HH
  if (idx >= (size_t)NT * HH) return;
  int j = (int)(idx % HH);
  size_t nt = idx / HH;
  int t = (int)(nt % TT);
  size_t n = nt / TT;
  float a = bias[j], g = bias[j + HH];
#pragma unroll
  for (int dt = 0; dt < 3; ++dt) {
    int tt = t + dt - 1;
    if (tt < 0 || tt >= TT) continue;
    const float* xin = df + (n * TT + tt) * FF;
    const float* wa = w + (size_t)j * (FF * 3) + dt;
    const float* wg = w + (size_t)(j + HH) * (FF * 3) + dt;
#pragma unroll
    for (int f = 0; f < FF; ++f) {
      a += wa[f * 3] * xin[f];
      g += wg[f * 3] * xin[f];
    }
  }
  float out = a * (1.f / (1.f + expf(-g)));
  x32[idx] = out;
  xh[idx] = (_Float16)out;
}

// ---------------- generic WMMA GEMM: C = act(A[MxK] * B[KxN] + bias) ----------------
// A: f16 row-major lda=K. Bp: f16 packed column-major [N][K]. Wave computes a 16x64
// output strip (4 N-tiles) so one A fetch feeds 4 WMMAs. Requires N%64==0, K%32==0.
__global__ __launch_bounds__(128)
void k_gemm(const _Float16* __restrict__ A, const _Float16* __restrict__ Bp,
            const float* __restrict__ bias,
            float* __restrict__ Cf, _Float16* __restrict__ Ch,
            int M, int N, int K, int ldc, int act) {
  int lane = threadIdx.x & 31;
  int wave = threadIdx.x >> 5;
  int sel = lane >> 4;          // half-wave select
  int r = lane & 15;
  int n4 = N >> 6;              // groups of 4 n-tiles
  int tiles = (M >> 4) * n4;
  int tileId = blockIdx.x * 4 + wave;
  if (tileId >= tiles) return;
  int tm = tileId / n4, t4 = tileId % n4;

  const _Float16* arow = A + (size_t)(tm * 16 + r) * K;
  const _Float16* bcol = Bp + (size_t)(t4 * 64 + r) * K;
  const size_t bstep = (size_t)16 * K;
  v8f acc0 = {}, acc1 = {}, acc2 = {}, acc3 = {};
  for (int k0 = 0; k0 < K; k0 += 32) {
    v16h a;
    ((v8h*)&a)[0] = *(const v8h*)(arow + k0 + 16 * sel);
    ((v8h*)&a)[1] = *(const v8h*)(arow + k0 + 16 * sel + 8);
    const _Float16* bp = bcol + k0 + 16 * sel;
    v16h b0, b1, b2, b3;
    ((v8h*)&b0)[0] = *(const v8h*)(bp);
    ((v8h*)&b0)[1] = *(const v8h*)(bp + 8);
    ((v8h*)&b1)[0] = *(const v8h*)(bp + bstep);
    ((v8h*)&b1)[1] = *(const v8h*)(bp + bstep + 8);
    ((v8h*)&b2)[0] = *(const v8h*)(bp + 2 * bstep);
    ((v8h*)&b2)[1] = *(const v8h*)(bp + 2 * bstep + 8);
    ((v8h*)&b3)[0] = *(const v8h*)(bp + 3 * bstep);
    ((v8h*)&b3)[1] = *(const v8h*)(bp + 3 * bstep + 8);
    acc0 = wmma_f16(a, b0, acc0);
    acc1 = wmma_f16(a, b1, acc1);
    acc2 = wmma_f16(a, b2, acc2);
    acc3 = wmma_f16(a, b3, acc3);
  }
#pragma unroll
  for (int q = 0; q < 4; ++q) {
    v8f acc = (q == 0) ? acc0 : (q == 1) ? acc1 : (q == 2) ? acc2 : acc3;
    int col = t4 * 64 + q * 16 + r;
    float bv = bias ? bias[col] : 0.f;
#pragma unroll
    for (int j = 0; j < 8; ++j) {
      int row = tm * 16 + j + 8 * sel;
      float v = acc[j] + bv;
      if (act == 1) v = fmaxf(v, 0.f);
      else if (act == 2) v = v / (1.f + expf(-v));
      else if (act == 3) v = tanhf(v);
      size_t o = (size_t)row * ldc + col;
      if (Cf) Cf[o] = v;
      if (Ch) Ch[o] = (_Float16)v;
    }
  }
}

static inline void gemm(hipStream_t st, const _Float16* A, const _Float16* Bp,
                        const float* bias, float* Cf, _Float16* Ch,
                        int M, int N, int K, int ldc, int act) {
  int tiles = (M / 16) * (N / 64);
  k_gemm<<<(tiles + 3) / 4, 128, 0, st>>>(A, Bp, bias, Cf, Ch, M, N, K, ldc, act);
}

// ---------------- fused attention per (n, head): WMMA scores + softmax + WMMA PV ----
__global__ __launch_bounds__(128)
void k_attn(const _Float16* __restrict__ qkvh, _Float16* __restrict__ oh) {
  __shared__ float    s_s[64 * 64];
  __shared__ _Float16 s_p[64 * 64];
  __shared__ _Float16 s_vt[32 * 64];
  int n = blockIdx.x >> 2, hdix = blockIdx.x & 3;
  int tid = threadIdx.x;
  int wave = tid >> 5, lane = tid & 31;
  int sel = lane >> 4, r = lane & 15;
  const _Float16* base = qkvh + (size_t)n * TT * 384;
  const _Float16* qp = base + hdix * 32;
  const _Float16* kp = base + 128 + hdix * 32;
  const _Float16* vp = base + 256 + hdix * 32;

  for (int idx = tid; idx < 32 * 64; idx += 128) {
    int hd = idx >> 6, t = idx & 63;
    s_vt[hd * 64 + t] = (t < TT) ? vp[(size_t)t * 384 + hd] : (_Float16)0.f;
  }

  int tm = wave;
  int tq = tm * 16 + r; if (tq > TT - 1) tq = TT - 1;     // clamp pad rows (masked later)
  const _Float16* arow = qp + (size_t)tq * 384;
  v16h qa;
  ((v8h*)&qa)[0] = *(const v8h*)(arow + 16 * sel);
  ((v8h*)&qa)[1] = *(const v8h*)(arow + 16 * sel + 8);
  const float scale = 0.17677669529663689f;               // 1/sqrt(32)
  for (int tn = 0; tn < 4; ++tn) {
    int tk = tn * 16 + r; if (tk > TT - 1) tk = TT - 1;
    const _Float16* bcol = kp + (size_t)tk * 384;         // k row-major == packed B for q.k^T
    v16h kb;
    ((v8h*)&kb)[0] = *(const v8h*)(bcol + 16 * sel);
    ((v8h*)&kb)[1] = *(const v8h*)(bcol + 16 * sel + 8);
    v8f acc = {};
    acc = wmma_f16(qa, kb, acc);
#pragma unroll
    for (int j = 0; j < 8; ++j)
      s_s[(tm * 16 + j + 8 * sel) * 64 + tn * 16 + r] = acc[j] * scale;
  }
  __syncthreads();

  if (tid < 64) {
    int row = tid;
    if (row < TT) {
      float m = -1e30f;
      for (int j = 0; j < TT; ++j) m = fmaxf(m, s_s[row * 64 + j]);
      float Z = 0.f;
      for (int j = 0; j < TT; ++j) Z += expf(s_s[row * 64 + j] - m);
      float inv = 1.f / Z;
      for (int j = 0; j < 64; ++j)
        s_p[row * 64 + j] = (j < TT) ? (_Float16)(expf(s_s[row * 64 + j] - m) * inv)
                                     : (_Float16)0.f;
    } else {
      for (int j = 0; j < 64; ++j) s_p[row * 64 + j] = (_Float16)0.f;
    }
  }
  __syncthreads();

  const _Float16* prow = s_p + (tm * 16 + r) * 64;
  v16h pa0, pa1;
  ((v8h*)&pa0)[0] = *(const v8h*)(prow + 16 * sel);
  ((v8h*)&pa0)[1] = *(const v8h*)(prow + 16 * sel + 8);
  ((v8h*)&pa1)[0] = *(const v8h*)(prow + 32 + 16 * sel);
  ((v8h*)&pa1)[1] = *(const v8h*)(prow + 32 + 16 * sel + 8);
  for (int tn = 0; tn < 2; ++tn) {
    const _Float16* bcol = s_vt + (tn * 16 + r) * 64;
    v16h b0, b1;
    ((v8h*)&b0)[0] = *(const v8h*)(bcol + 16 * sel);
    ((v8h*)&b0)[1] = *(const v8h*)(bcol + 16 * sel + 8);
    ((v8h*)&b1)[0] = *(const v8h*)(bcol + 32 + 16 * sel);
    ((v8h*)&b1)[1] = *(const v8h*)(bcol + 32 + 16 * sel + 8);
    v8f acc = {};
    acc = wmma_f16(pa0, b0, acc);
    acc = wmma_f16(pa1, b1, acc);
#pragma unroll
    for (int j = 0; j < 8; ++j) {
      int t = tm * 16 + j + 8 * sel;
      if (t < TT)
        oh[((size_t)n * TT + t) * HH + hdix * 32 + tn * 16 + r] = (_Float16)acc[j];
    }
  }
}

// ---------------- LayerNorm (wave per row): y = LN(a + res) * g + b ----------------
__global__ void k_ln(const float* __restrict__ a, const float* __restrict__ res,
                     const float* __restrict__ g, const float* __restrict__ b,
                     float* __restrict__ outf, _Float16* __restrict__ outh, int M, int D) {
  int row = blockIdx.x * 4 + (threadIdx.x >> 5);
  int lane = threadIdx.x & 31;
  if (row >= M) return;
  const float* ar = a + (size_t)row * D;
  const float* rr = res ? res + (size_t)row * D : nullptr;
  float s = 0.f, s2 = 0.f;
  for (int j = lane; j < D; j += 32) {
    float v = ar[j] + (rr ? rr[j] : 0.f);
    s += v; s2 += v * v;
  }
  s = wave_red(s); s2 = wave_red(s2);
  float mean = s / D;
  float var = s2 / D - mean * mean;
  float inv = rsqrtf(var + 1e-5f);
  for (int j = lane; j < D; j += 32) {
    float v = ar[j] + (rr ? rr[j] : 0.f);
    float y = (v - mean) * inv * g[j] + b[j];
    if (outf) outf[(size_t)row * D + j] = y;
    if (outh) outh[(size_t)row * D + j] = (_Float16)y;
  }
}

// ---------------- attention pooling ----------------
__global__ void k_pool(const float* __restrict__ tb, const float* __restrict__ w2,
                       const float* __restrict__ b2, const float* __restrict__ x,
                       float* __restrict__ hd) {
  __shared__ float sc[TT];
  int n = blockIdx.x, tid = threadIdx.x;       // 128 threads
  if (tid < TT) {
    const float* row = tb + ((size_t)n * TT + tid) * HH;
    float s = b2[0];
    for (int j = 0; j < HH; ++j) s += row[j] * w2[j];
    sc[tid] = s;
  }
  __syncthreads();
  if (tid == 0) {
    float m = -1e30f;
    for (int t = 0; t < TT; ++t) m = fmaxf(m, sc[t]);
    float Z = 0.f;
    for (int t = 0; t < TT; ++t) { sc[t] = expf(sc[t] - m); Z += sc[t]; }
    float inv = 1.f / Z;
    for (int t = 0; t < TT; ++t) sc[t] *= inv;
  }
  __syncthreads();
  float acc = 0.f;
  for (int t = 0; t < TT; ++t) acc += sc[t] * x[((size_t)n * TT + t) * HH + tid];
  hd[(size_t)n * HH + tid] = acc;
}

// ---------------- industry embedding gather (padding row never selected) ----------
__global__ void k_gather_emb(const float* __restrict__ emb, const int* __restrict__ ind,
                             _Float16* __restrict__ out) {
  int idx = blockIdx.x * 256 + threadIdx.x;
  if (idx >= NN * EMB) return;
  int n = idx / EMB, e = idx % EMB;
  out[idx] = (_Float16)emb[(size_t)ind[n] * EMB + e];
}

// ---------------- gate fuse (wave per row) ----------------
__global__ void k_gate(const float* __restrict__ hd, const float* __restrict__ hi,
                       const float* __restrict__ gw, const float* __restrict__ gb,
                       float* __restrict__ hp) {
  int n = blockIdx.x * 4 + (threadIdx.x >> 5);
  int lane = threadIdx.x & 31;
  if (n >= NN) return;
  float p = 0.f;
  for (int j = lane; j < HH; j += 32)
    p += hd[(size_t)n * HH + j] * gw[j] + hi[(size_t)n * HH + j] * gw[HH + j];
  p = wave_red(p);
  float g = 1.f / (1.f + expf(-(p + gb[0])));
  for (int j = lane; j < HH; j += 32)
    hp[(size_t)n * HH + j] = g * hd[(size_t)n * HH + j] + (1.f - g) * hi[(size_t)n * HH + j];
}

// ---------------- FiLM ----------------
__global__ void k_film(const float* __restrict__ hp, const float* __restrict__ gg,
                       const float* __restrict__ bb,
                       float* __restrict__ hctx, float* __restrict__ h32,
                       _Float16* __restrict__ h16) {
  int idx = blockIdx.x * 256 + threadIdx.x;
  if (idx >= NN * HH) return;
  float v = hp[idx] * (1.f + gg[idx]) + bb[idx];
  hctx[idx] = v; h32[idx] = v; h16[idx] = (_Float16)v;
}

// ---------------- industry grouping (counting sort) ----------------
__global__ void k_grp_init(int* counts, int* cursor) {
  int i = threadIdx.x;
  if (i < NIND) { counts[i] = 0; cursor[i] = 0; }
}
__global__ void k_hist(const int* __restrict__ ind, int* counts) {
  int n = blockIdx.x * 256 + threadIdx.x;
  if (n < NN) atomicAdd(&counts[ind[n]], 1);
}
__global__ void k_scan(const int* counts, int* offs) {
  if (threadIdx.x == 0) {
    int s = 0;
    for (int i = 0; i < NIND; ++i) { offs[i] = s; s += counts[i]; }
    offs[NIND] = s;
  }
}
__global__ void k_scatter(const int* __restrict__ ind, const int* __restrict__ offs,
                          int* cursor, int* order) {
  int n = blockIdx.x * 256 + threadIdx.x;
  if (n < NN) {
    int g = ind[n];
    int pos = offs[g] + atomicAdd(&cursor[g], 1);
    order[pos] = n;
  }
}

// ---------------- GAT: f1 = Wh@a1, f2 = Wh@a2 (wave per row) ----------------
__global__ void k_f12(const float* __restrict__ Wh, const float* __restrict__ att,
                      float* __restrict__ f1, float* __restrict__ f2) {
  int n = blockIdx.x * 4 + (threadIdx.x >> 5);
  int lane = threadIdx.x & 31;
  if (n >= NN) return;
  float p1 = 0.f, p2 = 0.f;
  for (int j = lane; j < HH; j += 32) {
    float w = Wh[(size_t)n * HH + j];
    p1 += w * att[j];
    p2 += w * att[HH + j];
  }
  p1 = wave_red(p1); p2 = wave_red(p2);
  if (lane == 0) { f1[n] = p1; f2[n] = p2; }
}

// ---------------- GAT per-industry dense attention ----------------
__global__ void k_gat_group(const int* __restrict__ order, const int* __restrict__ offs,
                            const float* __restrict__ f1, const float* __restrict__ f2,
                            const float* __restrict__ att, const float* __restrict__ Wh,
                            float* __restrict__ out) {
  int g = blockIdx.x;
  int s = offs[g], e = offs[g + 1];
  int G = e - s;
  float ae = att[2 * HH];
  for (int ii = s + (int)threadIdx.x; ii < e; ii += blockDim.x) {
    int i = order[ii];
    float* orow = out + (size_t)i * HH;
    if (G <= 1) {
      for (int c = 0; c < HH; ++c) orow[c] = 0.f;   // singleton groups skipped
      continue;
    }
    float fi = f1[i];
    float m = -1e30f;
    for (int jj = s; jj < e; ++jj) {
      float v = fi + f2[order[jj]] + ae;
      v = v > 0.f ? v : 0.1f * v;
      m = fmaxf(m, v);
    }
    float Z = 0.f;
    for (int jj = s; jj < e; ++jj) {
      float v = fi + f2[order[jj]] + ae;
      v = v > 0.f ? v : 0.1f * v;
      Z += expf(v - m);
    }
    float invZ = 1.f / Z;
    for (int c0 = 0; c0 < HH; c0 += 8) {
      float acc[8] = {0.f, 0.f, 0.f, 0.f, 0.f, 0.f, 0.f, 0.f};
      for (int jj = s; jj < e; ++jj) {
        int j = order[jj];
        float v = fi + f2[j] + ae;
        v = v > 0.f ? v : 0.1f * v;
        float p = expf(v - m) * invZ;
        const float* wr = Wh + (size_t)j * HH + c0;
#pragma unroll
        for (int c = 0; c < 8; ++c) acc[c] += p * wr[c];
      }
#pragma unroll
      for (int c = 0; c < 8; ++c) orow[c0 + c] = acc[c];
    }
  }
}

// ---------------- head concat + final dot ----------------
__global__ void k_concat(const float* __restrict__ hp, const float* __restrict__ hc,
                         const float* __restrict__ h, float* __restrict__ z) {
  int idx = blockIdx.x * 256 + threadIdx.x;
  if (idx >= NN * 3 * HH) return;
  int n = idx / (3 * HH), j = idx % (3 * HH);
  float v = (j < HH) ? hp[(size_t)n * HH + j]
          : (j < 2 * HH) ? hc[(size_t)n * HH + (j - HH)]
                         : h[(size_t)n * HH + (j - 2 * HH)];
  z[idx] = v;
}
__global__ void k_head_out(const float* __restrict__ z2, const float* __restrict__ w3,
                           const float* __restrict__ b3, float* __restrict__ out) {
  int n = blockIdx.x * 256 + threadIdx.x;
  if (n >= NN) return;
  float s = b3[0];
  for (int j = 0; j < 64; ++j) s += z2[(size_t)n * 64 + j] * w3[j];
  out[n] = s;
}

// =====================================================================
extern "C" void kernel_launch(void* const* d_in, const int* in_sizes, int n_in,
                              void* d_out, int out_size, void* d_ws, size_t ws_size,
                              hipStream_t stream) {
  (void)in_sizes; (void)n_in; (void)out_size; (void)ws_size;
  // ---- inputs (setup_inputs dict order, params flattened depth-first) ----
  const float* daily   = (const float*)d_in[0];
  const int*   ind_id  = (const int*)d_in[1];
  const float* ctx     = (const float*)d_in[2];
  const float* conv_w  = (const float*)d_in[3];
  const float* conv_b  = (const float*)d_in[4];
  const float* wq = (const float*)d_in[5];  const float* bq = (const float*)d_in[6];
  const float* wk = (const float*)d_in[7];  const float* bk = (const float*)d_in[8];
  const float* wv = (const float*)d_in[9];  const float* bv = (const float*)d_in[10];
  const float* wo = (const float*)d_in[11]; const float* bo = (const float*)d_in[12];
  const float* ln1g = (const float*)d_in[13]; const float* ln1b = (const float*)d_in[14];
  const float* w1 = (const float*)d_in[15]; const float* b1 = (const float*)d_in[16];
  const float* w2 = (const float*)d_in[17]; const float* b2 = (const float*)d_in[18];
  const float* ln2g = (const float*)d_in[19]; const float* ln2b = (const float*)d_in[20];
  const float* pw1 = (const float*)d_in[21]; const float* pb1 = (const float*)d_in[22];
  const float* pw2 = (const float*)d_in[23]; const float* pb2 = (const float*)d_in[24];
  const float* emb = (const float*)d_in[25];
  const float* ipw = (const float*)d_in[26]; const float* ipb = (const float*)d_in[27];
  const float* gw  = (const float*)d_in[28]; const float* gb  = (const float*)d_in[29];
  const float* cng = (const float*)d_in[30]; const float* cnb = (const float*)d_in[31];
  const float* fgw = (const float*)d_in[32]; const float* fgb = (const float*)d_in[33];
  const float* fbw = (const float*)d_in[34]; const float* fbb = (const float*)d_in[35];
  const float* gatW[2]   = {(const float*)d_in[36], (const float*)d_in[40]};
  const float* gatAtt[2] = {(const float*)d_in[37], (const float*)d_in[41]};
  const float* gatNg[2]  = {(const float*)d_in[38], (const float*)d_in[42]};
  const float* gatNb[2]  = {(const float*)d_in[39], (const float*)d_in[43]};
  const float* hlng = (const float*)d_in[44]; const float* hlnb = (const float*)d_in[45];
  const float* hw1 = (const float*)d_in[46]; const float* hb1 = (const float*)d_in[47];
  const float* hw2 = (const float*)d_in[48]; const float* hb2 = (const float*)d_in[49];
  const float* hw3 = (const float*)d_in[50]; const float* hb3 = (const float*)d_in[51];
  float* outp = (float*)d_out;

  // ---- workspace bump allocator ----
  char* wsB = (char*)d_ws;
  size_t off = 0;
  auto alloc = [&](size_t bytes) -> void* {
    void* p = wsB + off;
    off += (bytes + 255) & ~(size_t)255;
    return p;
  };
  float*     x32  = (float*)alloc((size_t)NT * HH * 4);
  float*     t32  = (float*)alloc((size_t)NT * HH * 4);
  _Float16*  xh   = (_Float16*)alloc((size_t)NT * HH * 2);
  _Float16*  th   = (_Float16*)alloc((size_t)NT * 512 * 2);  // qkv(384) / ffn-mid(512)
  _Float16*  oh   = (_Float16*)alloc((size_t)NT * HH * 2);
  float* hd32   = (float*)alloc((size_t)NN * HH * 4);
  float* hind32 = (float*)alloc((size_t)NN * HH * 4);
  float* hpr32  = (float*)alloc((size_t)NN * HH * 4);
  float* gg32   = (float*)alloc((size_t)NN * HH * 4);
  float* bb32   = (float*)alloc((size_t)NN * HH * 4);
  float* hctx32 = (float*)alloc((size_t)NN * HH * 4);
  float* h32    = (float*)alloc((size_t)NN * HH * 4);
  float* Wh32   = (float*)alloc((size_t)NN * HH * 4);
  float* gout32 = (float*)alloc((size_t)NN * HH * 4);
  _Float16* h16    = (_Float16*)alloc((size_t)NN * HH * 2);
  _Float16* ctxnh  = (_Float16*)alloc((size_t)NN * CTX * 2);
  _Float16* aembh  = (_Float16*)alloc((size_t)NN * EMB * 2);
  float* f1 = (float*)alloc((size_t)NN * 4);
  float* f2 = (float*)alloc((size_t)NN * 4);
  float*    z32 = (float*)alloc((size_t)NN * 384 * 4);
  _Float16* zh  = (_Float16*)alloc((size_t)NN * 384 * 2);
  _Float16* z1h = (_Float16*)alloc((size_t)NN * 192 * 2);
  float*    z2f = (float*)alloc((size_t)NN * 64 * 4);
  int* counts = (int*)alloc((NIND + 1) * 4);
  int* offs   = (int*)alloc((NIND + 1) * 4);
  int* cursor = (int*)alloc((NIND + 1) * 4);
  int* order  = (int*)alloc((size_t)NN * 4);
  // packed f16 weights
  _Float16* wq_p = (_Float16*)alloc(HH * HH * 2);
  _Float16* wk_p = (_Float16*)alloc(HH * HH * 2);
  _Float16* wv_p = (_Float16*)alloc(HH * HH * 2);
  _Float16* wo_p = (_Float16*)alloc(HH * HH * 2);
  _Float16* w1_p = (_Float16*)alloc(HH * 512 * 2);
  _Float16* w2_p = (_Float16*)alloc(512 * HH * 2);
  _Float16* pw1_p = (_Float16*)alloc(HH * HH * 2);
  _Float16* ipw_p = (_Float16*)alloc(EMB * HH * 2);
  _Float16* fgw_p = (_Float16*)alloc(CTX * HH * 2);
  _Float16* fbw_p = (_Float16*)alloc(CTX * HH * 2);
  _Float16* gat_p[2] = {(_Float16*)alloc(HH * HH * 2), (_Float16*)alloc(HH * HH * 2)};
  _Float16* hw1_p = (_Float16*)alloc(384 * 192 * 2);
  _Float16* hw2_p = (_Float16*)alloc(192 * 64 * 2);

  auto pack = [&](const float* w, _Float16* dst, int K, int N) {
    k_pack<<<(K * N + 255) / 256, 256, 0, stream>>>(w, dst, K, N);
  };
  pack(wq, wq_p, HH, HH); pack(wk, wk_p, HH, HH); pack(wv, wv_p, HH, HH);
  pack(wo, wo_p, HH, HH); pack(w1, w1_p, HH, 512); pack(w2, w2_p, 512, HH);
  pack(pw1, pw1_p, HH, HH); pack(ipw, ipw_p, EMB, HH);
  pack(fgw, fgw_p, CTX, HH); pack(fbw, fbw_p, CTX, HH);
  pack(gatW[0], gat_p[0], HH, HH); pack(gatW[1], gat_p[1], HH, HH);
  pack(hw1, hw1_p, 384, 192); pack(hw2, hw2_p, 192, 64);

  // ---- conv + GLU ----
  k_conv_glu<<<(NT * HH + 255) / 256, 256, 0, stream>>>(daily, conv_w, conv_b, x32, xh);

  // ---- transformer layer ----
  gemm(stream, xh, wq_p, bq, nullptr, th + 0,   NT, HH, HH, 384, 0);
  gemm(stream, xh, wk_p, bk, nullptr, th + 128, NT, HH, HH, 384, 0);
  gemm(stream, xh, wv_p, bv, nullptr, th + 256, NT, HH, HH, 384, 0);
  k_attn<<<NN * NHEAD, 128, 0, stream>>>(th, oh);
  gemm(stream, oh, wo_p, bo, t32, nullptr, NT, HH, HH, HH, 0);
  k_ln<<<(NT + 3) / 4, 128, 0, stream>>>(x32, t32, ln1g, ln1b, x32, xh, NT, HH);
  gemm(stream, xh, w1_p, b1, nullptr, th, NT, 512, HH, 512, 1);   // relu
  gemm(stream, th, w2_p, b2, t32, nullptr, NT, HH, 512, HH, 0);
  k_ln<<<(NT + 3) / 4, 128, 0, stream>>>(x32, t32, ln2g, ln2b, x32, xh, NT, HH);

  // ---- attention pooling ----
  gemm(stream, xh, pw1_p, pb1, t32, nullptr, NT, HH, HH, HH, 3);  // tanh
  k_pool<<<NN, 128, 0, stream>>>(t32, pw2, pb2, x32, hd32);

  // ---- industry embed + gate fuse ----
  k_gather_emb<<<(NN * EMB + 255) / 256, 256, 0, stream>>>(emb, ind_id, aembh);
  gemm(stream, aembh, ipw_p, ipb, hind32, nullptr, NN, HH, EMB, HH, 0);
  k_gate<<<(NN + 3) / 4, 128, 0, stream>>>(hd32, hind32, gw, gb, hpr32);

  // ---- FiLM ----
  k_ln<<<(NN + 3) / 4, 128, 0, stream>>>(ctx, nullptr, cng, cnb, nullptr, ctxnh, NN, CTX);
  gemm(stream, ctxnh, fgw_p, fgb, gg32, nullptr, NN, HH, CTX, HH, 0);
  gemm(stream, ctxnh, fbw_p, fbb, bb32, nullptr, NN, HH, CTX, HH, 0);
  k_film<<<(NN * HH + 255) / 256, 256, 0, stream>>>(hpr32, gg32, bb32, hctx32, h32, h16);

  // ---- GAT: group by industry, 2 layers ----
  k_grp_init<<<1, 256, 0, stream>>>(counts, cursor);
  k_hist<<<(NN + 255) / 256, 256, 0, stream>>>(ind_id, counts);
  k_scan<<<1, 64, 0, stream>>>(counts, offs);
  k_scatter<<<(NN + 255) / 256, 256, 0, stream>>>(ind_id, offs, cursor, order);
  for (int l = 0; l < 2; ++l) {
    gemm(stream, h16, gat_p[l], nullptr, Wh32, nullptr, NN, HH, HH, HH, 0);
    k_f12<<<(NN + 3) / 4, 128, 0, stream>>>(Wh32, gatAtt[l], f1, f2);
    k_gat_group<<<NIND, 256, 0, stream>>>(order, offs, f1, f2, gatAtt[l], Wh32, gout32);
    k_ln<<<(NN + 3) / 4, 128, 0, stream>>>(gout32, h32, gatNg[l], gatNb[l], h32, h16, NN, HH);
  }

  // ---- head ----
  k_concat<<<(NN * 384 + 255) / 256, 256, 0, stream>>>(hpr32, hctx32, h32, z32);
  k_ln<<<(NN + 3) / 4, 128, 0, stream>>>(z32, nullptr, hlng, hlnb, nullptr, zh, NN, 384);
  gemm(stream, zh, hw1_p, hb1, nullptr, z1h, NN, 192, 384, 192, 2);  // silu
  gemm(stream, z1h, hw2_p, hb2, z2f, nullptr, NN, 64, 192, 64, 2);   // silu
  k_head_out<<<(NN + 255) / 256, 256, 0, stream>>>(z2f, hw3, hb3, outp);
}
